// ClusterMemory_48404281426351
// MI455X (gfx1250) — compile-verified, matrix operand order checked by reading, not run on hardware
//
#include <hip/hip_runtime.h>
#include <math.h>
#include <stdint.h>

// ---------- problem constants (from reference) ----------
#define BB    512          // batch
#define DD    768          // feature dim
#define PP    128          // patch tokens
#define NINST 65536        // instance memory
#define CC    4096         // clusters
#define NSTRIPE 128        // 65536 / 512 cols per stripe
#define STRIPE  512
#define INV_T 20.0f        // 1 / 0.05

typedef __attribute__((ext_vector_type(16))) _Float16 v16h;
typedef __attribute__((ext_vector_type(8)))  _Float16 v8h;
typedef __attribute__((ext_vector_type(8)))  float    v8f;
typedef __attribute__((ext_vector_type(4)))  unsigned int v4u;
typedef __attribute__((ext_vector_type(8)))  int      v8i;
typedef __attribute__((ext_vector_type(4)))  int      v4i;

#define DEVINL __device__ __forceinline__
#define FINF  __builtin_huge_valf()

#if defined(__has_builtin)
#if __has_builtin(__builtin_amdgcn_tensor_load_to_lds) && __has_builtin(__builtin_amdgcn_s_wait_tensorcnt)
#define USE_TDM 1
#endif
#endif

// ---------------- wave32 reductions ----------------
DEVINL float wredSum(float v) {
#pragma unroll
  for (int j = 16; j > 0; j >>= 1) v += __shfl_xor(v, j, 32);
  return v;
}
DEVINL float wredMin(float v) {
#pragma unroll
  for (int j = 16; j > 0; j >>= 1) v = fminf(v, __shfl_xor(v, j, 32));
  return v;
}
DEVINL float wredMax(float v) {
#pragma unroll
  for (int j = 16; j > 0; j >>= 1) v = fmaxf(v, __shfl_xor(v, j, 32));
  return v;
}

// ------- 64-element bitonic sort (2 regs/lane, idx = lane + 32*r), descending -------
DEVINL void cmpex64(float& v0, float& v1, int lane, int k, int j) {
  if (j == 32) {                 // partner is the other register, same lane (k==64 stage)
    float hi = fmaxf(v0, v1), lo = fminf(v0, v1);
    v0 = hi; v1 = lo;
  } else {
    float o0 = __shfl_xor(v0, j, 32);
    float o1 = __shfl_xor(v1, j, 32);
    bool low = ((lane & j) == 0);
    bool m0  = ((lane & k) == 0);
    bool m1  = (((lane + 32) & k) == 0);
    v0 = (m0 == low) ? fmaxf(v0, o0) : fminf(v0, o0);
    v1 = (m1 == low) ? fmaxf(v1, o1) : fminf(v1, o1);
  }
}
DEVINL void sort64_desc(float& v0, float& v1, int lane) {
  for (int k = 2; k <= 64; k <<= 1)
    for (int j = k >> 1; j > 0; j >>= 1)
      cmpex64(v0, v1, lane, k, j);
}
DEVINL void merge64_desc(float& v0, float& v1, int lane) {
  for (int j = 32; j > 0; j >>= 1) cmpex64(v0, v1, lane, 64, j);
}

// ---------------- WMMA fragment loaders ----------------
// A frag (16x32 f16): lanes 0-15 -> K 0..7 & 16..23, lanes 16-31 -> K 8..15 & 24..31
DEVINL v16h loadAfrag(const _Float16* rowPtr, int kb, int khalf) {
  v8h a0 = *(const v8h*)(rowPtr + kb + khalf);
  v8h a1 = *(const v8h*)(rowPtr + kb + 16 + khalf);
  v16h r;
#pragma unroll
  for (int i = 0; i < 8; ++i) { r[i] = a0[i]; r[8 + i] = a1[i]; }
  return r;
}
// B frag (32x16 f16) from f32 data where each "row" of length rowstride is one B column
DEVINL v16h loadBfragF32(const float* colPtr, int kb, int khalf) {
  float4 a = *(const float4*)(colPtr + kb + khalf);
  float4 b = *(const float4*)(colPtr + kb + khalf + 4);
  float4 c = *(const float4*)(colPtr + kb + 16 + khalf);
  float4 d = *(const float4*)(colPtr + kb + 16 + khalf + 4);
  v16h r;
  r[0] = (_Float16)a.x; r[1] = (_Float16)a.y; r[2]  = (_Float16)a.z; r[3]  = (_Float16)a.w;
  r[4] = (_Float16)b.x; r[5] = (_Float16)b.y; r[6]  = (_Float16)b.z; r[7]  = (_Float16)b.w;
  r[8] = (_Float16)c.x; r[9] = (_Float16)c.y; r[10] = (_Float16)c.z; r[11] = (_Float16)c.w;
  r[12]= (_Float16)d.x; r[13]= (_Float16)d.y; r[14] = (_Float16)d.z; r[15] = (_Float16)d.w;
  return r;
}

#ifdef USE_TDM
// TDM: DMA a (rows=64) x (cols=32 f32) tile -> LDS, row stride DD elements.
// D# per CDNA5 ISA (2D tensor, groups 2/3 zero).
// This toolchain exposes the 6-arg builtin: (v4u g0, v8i g1, v4i, v4i, v8i, i32 cpol).
DEVINL void tdm_load_slab(const float* gsrc, void* ldsdst) {
  uint64_t ga  = (uint64_t)(uintptr_t)gsrc;
  uint32_t lds = (uint32_t)(uintptr_t)ldsdst;
  v4u g0 = { 1u,                                   // count=1, user descriptor
             lds,                                  // lds_addr
             (unsigned)(ga & 0xffffffffu),         // global_addr[31:0]
             (unsigned)((ga >> 32) & 0x01ffffffu) | (2u << 30) }; // [56:32] | type=2
  v8i g1 = { (int)(2u << 16),                      // data_size = 4B, mask=0
             (int)(32u << 16),                     // tensor_dim0 = 32
             (int)(64u << 16),                     // tensor_dim1 = 64
             (int)(32u << 16),                     // tile_dim0 = 32
             (int)64,                              // tile_dim1 = 64, tile_dim2 = 0
             (int)DD,                              // tensor_dim0_stride = 768
             0, 0 };
  v4i gz4 = { 0, 0, 0, 0 };
  v8i gz8 = { 0, 0, 0, 0, 0, 0, 0, 0 };
  __builtin_amdgcn_tensor_load_to_lds(g0, g1, gz4, gz4, gz8, 0);
}
#endif

// ================= Kernel 1: row-normalize x -> f16 =================
__global__ void __launch_bounds__(256) norm_kernel(const float* __restrict__ x,
                                                   _Float16* __restrict__ xnh) {
  __shared__ float red[256];
  int row = blockIdx.x, tid = threadIdx.x;
  const float* xr = x + (size_t)row * DD;
  float s = 0.f;
  for (int i = tid; i < DD; i += 256) { float v = xr[i]; s += v * v; }
  red[tid] = s; __syncthreads();
  for (int st = 128; st > 0; st >>= 1) { if (tid < st) red[tid] += red[tid + st]; __syncthreads(); }
  float inv = rsqrtf(red[0]);
  _Float16* orow = xnh + (size_t)row * DD;
  for (int i = tid; i < DD; i += 256) orow[i] = (_Float16)(xr[i] * inv);
}

// ================= Kernel 2: patch loss term per row =================
__global__ void __launch_bounds__(256) patch_kernel(const float* __restrict__ cls,
                                                    const float* __restrict__ tokens,
                                                    float* __restrict__ patchTerm) {
  __shared__ float clsS[DD];
  __shared__ float pm[PP];
  int b = blockIdx.x, tid = threadIdx.x;
  int wave = tid >> 5, lane = tid & 31;
  const float* cr = cls + (size_t)b * DD;
  for (int i = tid; i < DD; i += 256) clsS[i] = cr[i];
  __syncthreads();
  for (int pi = 0; pi < 16; ++pi) {
    int p = wave * 16 + pi;
    const float* tp = tokens + ((size_t)b * PP + p) * DD;
    float s = 0.f;
#pragma unroll
    for (int t = 0; t < DD / 32; ++t) { int k = lane + 32 * t; s += clsS[k] * tp[k]; }
    s = wredSum(s);
    if (lane == 0) pm[p] = s;
  }
  __syncthreads();
  // bitonic sort ascending (128 elems)
  for (int k = 2; k <= PP; k <<= 1)
    for (int j = k >> 1; j > 0; j >>= 1) {
      if (tid < 64) {
        int i = ((tid & ~(j - 1)) << 1) | (tid & (j - 1));
        int ixj = i | j;
        bool up = ((i & k) == 0);
        float a = pm[i], c = pm[ixj];
        if ((a > c) == up) { pm[i] = c; pm[ixj] = a; }
      }
      __syncthreads();
    }
  if (wave == 0) {
    float pp = pm[PP - 1];
    float e = __expf((pm[lane] - pp) * INV_T) + __expf((pm[lane + 32] - pp) * INV_T);
    e = wredSum(e);
    if (lane == 0) patchTerm[b] = logf(1.0f + e);
  }
}

// ===== Kernel 3: anchor GEMM (WMMA f16, TDM-staged B) + fused top-64 / pos-min =====
__global__ void __launch_bounds__(256) anchor_gemm_kernel(const _Float16* __restrict__ xnh,
                                                          const float* __restrict__ mem,
                                                          const int* __restrict__ labels,
                                                          const int* __restrict__ targets,
                                                          float* __restrict__ partMin,
                                                          float* __restrict__ partTop) {
  __shared__ float sS[64][65];     // 64x64 score tile (+pad)
  __shared__ float sTop[64][64];   // running per-row top-64, descending
  __shared__ float sMin[64];       // running per-row positive min
  __shared__ float sB[2][64 * 32]; // double-buffered B slab: 64 cols x 32 K (f32)
  int tid = threadIdx.x, wave = tid >> 5, lane = tid & 31;
  int rowBase = blockIdx.x * 64;           // 8 row blocks
  int stripe  = blockIdx.y;                // 128 stripes
  int scol0   = stripe * STRIPE;

  for (int i = tid; i < 64 * 64; i += 256) sTop[i >> 6][i & 63] = -FINF;
  for (int i = tid; i < 64; i += 256) sMin[i] = FINF;
  __syncthreads();

  int mtile = (wave & 3) * 16;             // 4 M tiles
  int cgrp  = (wave >> 2) * 32;            // 2 col groups of 32
  int khalf = (lane >> 4) << 3;            // 0 or 8
  const _Float16* aRow = xnh + (size_t)(rowBase + mtile + (lane & 15)) * DD;
  const int nKS = DD / 32;                 // 24 K-steps

  for (int ch = 0; ch < STRIPE / 64; ++ch) {
    int cb = scol0 + ch * 64;
    const float* memChunk = mem + (size_t)cb * DD;
#ifdef USE_TDM
    if (wave == 0) tdm_load_slab(memChunk, &sB[0][0]);   // prologue: K slab 0
#endif
    v8f acc0 = {}; v8f acc1 = {};
    for (int ks = 0; ks < nKS; ++ks) {
      int kb = ks * 32;
#ifdef USE_TDM
      if (wave == 0) {
        if (ks + 1 < nKS) {
          tdm_load_slab(memChunk + kb + 32, &sB[(ks + 1) & 1][0]);
          __builtin_amdgcn_s_wait_tensorcnt(1);   // current slab complete, next in flight
        } else {
          __builtin_amdgcn_s_wait_tensorcnt(0);
        }
      }
#else
      // fallback: cooperative stage of the K slab
      for (int i = tid; i < 64 * 32; i += 256)
        sB[ks & 1][i] = memChunk[(size_t)(i >> 5) * DD + kb + (i & 31)];
#endif
      __syncthreads();
      const float* bbase = &sB[ks & 1][0];
      v16h a  = loadAfrag(aRow, kb, khalf);
      v16h b0 = loadBfragF32(bbase + (size_t)(cgrp + (lane & 15)) * 32, 0, khalf);
      v16h b1 = loadBfragF32(bbase + (size_t)(cgrp + 16 + (lane & 15)) * 32, 0, khalf);
      acc0 = __builtin_amdgcn_wmma_f32_16x16x32_f16(false, a, false, b0, (short)0, acc0, false, false);
      acc1 = __builtin_amdgcn_wmma_f32_16x16x32_f16(false, a, false, b1, (short)0, acc1, false, false);
      __syncthreads();
    }
    // scatter C: VGPR i -> M = mtile + i (+8 for upper half-wave), N = cgrp + (lane&15)
    int mrow = mtile + ((lane >> 4) << 3);
    int ncol = cgrp + (lane & 15);
#pragma unroll
    for (int i = 0; i < 8; ++i) { sS[mrow + i][ncol] = acc0[i]; sS[mrow + i][ncol + 16] = acc1[i]; }
    __syncthreads();

    // selection: wave w owns rows w*8 .. w*8+7
    for (int rr = 0; rr < 8; ++rr) {
      int row = wave * 8 + rr;
      int tgt = targets[rowBase + row];
      float v0 = sS[row][lane], v1 = sS[row][lane + 32];
      int lb0 = labels[cb + lane], lb1 = labels[cb + lane + 32];
      bool p0 = (lb0 == tgt), p1 = (lb1 == tgt);
      // positive min
      float pmn = fminf(p0 ? v0 : FINF, p1 ? v1 : FINF);
      pmn = wredMin(pmn);
      if (lane == 0) sMin[row] = fminf(sMin[row], pmn);
      // negative top-64: skip sort if chunk can't beat current 64th-best
      float n0 = p0 ? -FINF : v0;
      float n1 = p1 ? -FINF : v1;
      float cmx = wredMax(fmaxf(n0, n1));
      float thr = sTop[row][63];
      if (cmx > thr) {
        sort64_desc(n0, n1, lane);
        float r0 = sTop[row][lane], r1 = sTop[row][lane + 32];
        float c0 = __shfl_xor(n1, 31, 32);   // chunk[63 - lane]
        float c1 = __shfl_xor(n0, 31, 32);   // chunk[63 - (lane+32)]
        float m0 = fmaxf(r0, c0), m1 = fmaxf(r1, c1);
        merge64_desc(m0, m1, lane);
        sTop[row][lane] = m0; sTop[row][lane + 32] = m1;
      }
    }
    __syncthreads();
  }
  // write partials
  for (int i = tid; i < 64 * 64; i += 256) {
    int row = i >> 6, c = i & 63;
    partTop[((size_t)(rowBase + row) * NSTRIPE + stripe) * 64 + c] = sTop[row][c];
  }
  for (int i = tid; i < 64; i += 256)
    partMin[(size_t)(rowBase + i) * NSTRIPE + stripe] = sMin[i];
}

// ===== Kernel 4: merge stripe partials -> exact top-k + anchor CE term =====
__global__ void __launch_bounds__(256) anchor_merge_kernel(const float* __restrict__ partMin,
                                                           const float* __restrict__ partTop,
                                                           const int* __restrict__ kp,
                                                           float* __restrict__ anchorTerm) {
  __shared__ float cand[NSTRIPE * 64];   // 8192 candidates
  __shared__ float rv[256];
  __shared__ int   ri[256];
  __shared__ float topv[64];
  int row = blockIdx.x, tid = threadIdx.x;
  for (int i = tid; i < NSTRIPE * 64; i += 256) cand[i] = partTop[(size_t)row * NSTRIPE * 64 + i];
  float lmin = FINF;
  for (int i = tid; i < NSTRIPE; i += 256) lmin = fminf(lmin, partMin[(size_t)row * NSTRIPE + i]);
  rv[tid] = lmin; __syncthreads();
  for (int s = 128; s > 0; s >>= 1) { if (tid < s) rv[tid] = fminf(rv[tid], rv[tid + s]); __syncthreads(); }
  float apos = rv[0]; __syncthreads();

  int k = *kp; if (k > 64) k = 64; if (k < 1) k = 1;
  for (int it = 0; it < k; ++it) {
    float bv = -FINF; int bi = 0;
    for (int i = tid; i < NSTRIPE * 64; i += 256)
      if (cand[i] > bv) { bv = cand[i]; bi = i; }
    rv[tid] = bv; ri[tid] = bi; __syncthreads();
    for (int s = 128; s > 0; s >>= 1) {
      if (tid < s) {
        if (rv[tid + s] > rv[tid] || (rv[tid + s] == rv[tid] && ri[tid + s] < ri[tid])) {
          rv[tid] = rv[tid + s]; ri[tid] = ri[tid + s];
        }
      }
      __syncthreads();
    }
    if (tid == 0) { topv[it] = rv[0]; cand[ri[0]] = -FINF; }
    __syncthreads();
  }
  if (tid == 0) {
    float mlog = fmaxf(apos, topv[0]) * INV_T;
    float sum = __expf(apos * INV_T - mlog);
    for (int i = 0; i < k; ++i) sum += __expf(topv[i] * INV_T - mlog);
    anchorTerm[row] = mlog + logf(sum) - apos * INV_T;
  }
}

// ===== Kernel 5: cluster logits (WMMA) + fused online-softmax CE =====
__global__ void __launch_bounds__(256) cluster_kernel(const _Float16* __restrict__ xnh,
                                                      const float* __restrict__ cf,
                                                      const int* __restrict__ targets,
                                                      float* __restrict__ clusterTerm) {
  __shared__ float sS[64][65];
  __shared__ float rM[64], rS[64], rT[64];
  int tid = threadIdx.x, wave = tid >> 5, lane = tid & 31;
  int rowBase = blockIdx.x * 64;           // 8 blocks
  for (int i = tid; i < 64; i += 256) { rM[i] = -FINF; rS[i] = 0.f; rT[i] = 0.f; }
  __syncthreads();

  int mtile = (wave & 3) * 16;
  int cgrp  = (wave >> 2) * 32;
  int khalf = (lane >> 4) << 3;
  const _Float16* aRow = xnh + (size_t)(rowBase + mtile + (lane & 15)) * DD;

  for (int ch = 0; ch < CC / 64; ++ch) {
    int cb = ch * 64;
    const float* bCol0 = cf + (size_t)(cb + cgrp + (lane & 15)) * DD;
    const float* bCol1 = bCol0 + (size_t)16 * DD;
    v8f acc0 = {}; v8f acc1 = {};
#pragma unroll 2
    for (int kb = 0; kb < DD; kb += 32) {
      v16h a  = loadAfrag(aRow, kb, khalf);
      v16h b0 = loadBfragF32(bCol0, kb, khalf);
      v16h b1 = loadBfragF32(bCol1, kb, khalf);
      acc0 = __builtin_amdgcn_wmma_f32_16x16x32_f16(false, a, false, b0, (short)0, acc0, false, false);
      acc1 = __builtin_amdgcn_wmma_f32_16x16x32_f16(false, a, false, b1, (short)0, acc1, false, false);
    }
    int mrow = mtile + ((lane >> 4) << 3);
    int ncol = cgrp + (lane & 15);
#pragma unroll
    for (int i = 0; i < 8; ++i) { sS[mrow + i][ncol] = acc0[i]; sS[mrow + i][ncol + 16] = acc1[i]; }
    __syncthreads();

    for (int rr = 0; rr < 8; ++rr) {
      int row = wave * 8 + rr;
      int tgt = targets[rowBase + row];
      float l0 = sS[row][lane] * INV_T;
      float l1 = sS[row][lane + 32] * INV_T;
      float cmax = wredMax(fmaxf(l0, l1));
      float oldM = rM[row];
      float nM = fmaxf(oldM, cmax);
      float e = __expf(l0 - nM) + __expf(l1 - nM);
      e = wredSum(e);
      if (lane == 0) { rS[row] = rS[row] * __expf(oldM - nM) + e; rM[row] = nM; }
      if (cb + lane == tgt)      rT[row] = l0;
      if (cb + lane + 32 == tgt) rT[row] = l1;
    }
    __syncthreads();
  }
  for (int i = tid; i < 64; i += 256)
    clusterTerm[rowBase + i] = rM[i] + logf(rS[i]) - rT[i];
}

// ===== Kernel 6: deterministic final reduction of 1536 per-row terms =====
__global__ void __launch_bounds__(256) final_reduce_kernel(const float* __restrict__ terms,
                                                           float* __restrict__ out) {
  __shared__ float red[256];
  int tid = threadIdx.x;
  float s = 0.f;
  for (int i = tid; i < 3 * BB; i += 256) s += terms[i];
  red[tid] = s; __syncthreads();
  for (int st = 128; st > 0; st >>= 1) { if (tid < st) red[tid] += red[tid + st]; __syncthreads(); }
  if (tid == 0) out[0] = red[0] / (float)BB;
}

// ================= host launcher =================
extern "C" void kernel_launch(void* const* d_in, const int* in_sizes, int n_in,
                              void* d_out, int out_size, void* d_ws, size_t ws_size,
                              hipStream_t stream) {
  (void)in_sizes; (void)n_in; (void)out_size; (void)ws_size;
  const float* x        = (const float*)d_in[0];
  const float* cls      = (const float*)d_in[1];
  const float* tokens   = (const float*)d_in[2];
  const float* mem      = (const float*)d_in[3];
  const int*   labels   = (const int*)d_in[4];
  const float* clusterF = (const float*)d_in[5];
  const int*   targets  = (const int*)d_in[6];
  // d_in[7] = indexes (unused by reference math)
  const int*   kp       = (const int*)d_in[8];
  float* out = (float*)d_out;

  char* ws = (char*)d_ws;
  _Float16* xnh    = (_Float16*)(ws);                       // 512*768*2  = 786432 B
  float* partMin   = (float*)(ws + 786432);                 // 512*128*4  = 262144 B
  float* partTop   = (float*)(ws + 1048576);                // 512*128*64*4 = 16 MiB
  float* terms     = (float*)(ws + 17825792);               // 1536 floats
  float* patchTerm   = terms;
  float* anchorTerm  = terms + BB;
  float* clusterTerm = terms + 2 * BB;

  norm_kernel<<<BB, 256, 0, stream>>>(x, xnh);
  patch_kernel<<<BB, 256, 0, stream>>>(cls, tokens, patchTerm);
  anchor_gemm_kernel<<<dim3(8, NSTRIPE), 256, 0, stream>>>(xnh, mem, labels, targets, partMin, partTop);
  anchor_merge_kernel<<<BB, 256, 0, stream>>>(partMin, partTop, kp, anchorTerm);
  cluster_kernel<<<8, 256, 0, stream>>>(xnh, clusterF, targets, clusterTerm);
  final_reduce_kernel<<<1, 256, 0, stream>>>(terms, out);
}
